// MultiHeadAttention_37520834298341
// MI455X (gfx1250) — compile-verified, matrix operand order checked by reading, not run on hardware
//
#include <hip/hip_runtime.h>
#include <hip/hip_bf16.h>

// ---------------------------------------------------------------------------
// MHA for B=4, S=2048, E=512, H=8, D=64 on gfx1250 (CDNA5, wave32, WMMA).
//   1) convert inputs f32->f16; transpose+convert weights f32->f16
//   2) WMMA GEMM (16x64 strip per wave) q,k -> [b,h,s,d] ; v -> [b,h,d,s]
//   3) flash-attention, 32 queries per wave, online softmax,
//      mask-before-scale per reference
//   4) WMMA GEMM output projection -> f32 d_out
// ---------------------------------------------------------------------------

typedef __attribute__((ext_vector_type(16))) _Float16 v16h;
typedef __attribute__((ext_vector_type(8)))  _Float16 v8h;
typedef __attribute__((ext_vector_type(8)))  float    v8f;

#define Bc 4
#define Sc 2048
#define Ec 512
#define Hc 8
#define Dc 64
#define HDc (Hc*Dc)     // 512
#define Mc (Bc*Sc)      // 8192

union V16H { v16h v; v8h h[2]; };

// Load one 16x32 f16 WMMA fragment (A or B operand) from row-major storage.
// CDNA5 16-bit operand layout: lanes 0-15 hold K 0-7 then 16-23,
// lanes 16-31 hold K 8-15 then 24-31; row/col = lane & 15 (caller passes row).
__device__ __forceinline__ v16h load_frag(const _Float16* __restrict__ base,
                                          int ld, int row, int k0, int lane) {
    int hi = (lane >> 4) & 1;
    const _Float16* p = base + (size_t)row * (size_t)ld + k0 + hi * 8;
    V16H u;
    u.h[0] = *(const v8h*)(p);
    u.h[1] = *(const v8h*)(p + 16);
    return u.v;
}

__device__ __forceinline__ v8f wmma_f16(v16h a, v16h b, v8f c) {
    return __builtin_amdgcn_wmma_f32_16x16x32_f16(false, a, false, b,
                                                  (short)0, c, false, false);
}

// ---------------------------------------------------------------------------
__global__ void k_f32_to_f16(const float* __restrict__ in,
                             _Float16* __restrict__ out, int n) {
    int i = blockIdx.x * blockDim.x + threadIdx.x;
    if (i < n) out[i] = (_Float16)in[i];
}

// in [R][C] f32 -> out [C][R] f16
__global__ void k_transpose_f16(const float* __restrict__ in,
                                _Float16* __restrict__ out, int R, int C) {
    int c = blockIdx.x * blockDim.x + threadIdx.x;
    int r = blockIdx.y;
    if (c < C) out[(size_t)c * R + r] = (_Float16)in[(size_t)r * C + c];
}

// ---------------------------------------------------------------------------
// C[M,N] = A[M,K]f16 * Wt[N,K]f16 + bias[N]
// Each wave computes a 16x64 strip: one shared A fragment, 4 accumulators.
// All 5 fragment loads for a k-step are issued before the 4-WMMA chain so the
// scheduler can use partial s_wait_loadcnt instead of full drains.
// mode 0: out16 as [b,h,s,d]   (q,k)
// mode 2: out16 as [b,h,d,s]   (v, transposed for P*V B-fragments)
// mode 3: out32 row-major [M,N] (final projection, f32)
__global__ void k_gemm_wmma(const _Float16* __restrict__ A,
                            const _Float16* __restrict__ Wt,
                            const float* __restrict__ bias,
                            _Float16* __restrict__ out16,
                            float* __restrict__ out32,
                            int M, int N, int K, int mode) {
    int wave = (blockIdx.x * blockDim.x + threadIdx.x) >> 5;
    int lane = threadIdx.x & 31;
    int groupsN = N >> 6;                 // 64-wide N groups
    int tm = wave / groupsN;
    int tg = wave - tm * groupsN;
    if (tm * 16 >= M) return;

    int l  = lane & 15;
    int hi = (lane >> 4) & 1;
    int arow = tm * 16 + l;

    v8f acc[4] = {{}, {}, {}, {}};
#pragma unroll 2
    for (int k0 = 0; k0 < K; k0 += 32) {
        v16h a = load_frag(A, K, arow, k0, lane);
        v16h b[4];
#pragma unroll
        for (int j = 0; j < 4; ++j)
            b[j] = load_frag(Wt, K, tg * 64 + j * 16 + l, k0, lane);
#pragma unroll
        for (int j = 0; j < 4; ++j)
            acc[j] = wmma_f16(a, b[j], acc[j]);
    }

#pragma unroll
    for (int j = 0; j < 4; ++j) {
        int n  = tg * 64 + j * 16 + l;
        float bv = bias[n];
        if (mode == 3) {
#pragma unroll
            for (int r = 0; r < 8; ++r) {
                int m = tm * 16 + r + hi * 8;
                out32[(size_t)m * N + n] = acc[j][r] + bv;
            }
        } else {
            int h = n >> 6, d = n & 63;
#pragma unroll
            for (int r = 0; r < 8; ++r) {
                int m  = tm * 16 + r + hi * 8;
                int bb = m >> 11, s = m & (Sc - 1);
                size_t idx;
                if (mode == 2) idx = ((((size_t)bb * Hc + h) * Dc + d) * Sc + s);
                else           idx = ((((size_t)bb * Hc + h) * Sc + s) * Dc + d);
                out16[idx] = (_Float16)(acc[j][r] + bv);
            }
        }
    }
}

// ---------------------------------------------------------------------------
// Flash attention: one wave per 32-query strip (2 q-tiles) of one (b,h).
// K/V fragments are loaded once per 32-key step and reused by both q-tiles.
// qh,kh: [B,H,S,D] f16 ; vt: [B,H,D,S] f16 ; mask: [B,S,S] i32
// o: [B,S,H*D] f16
__global__ void k_flash_attn(const _Float16* __restrict__ qh,
                             const _Float16* __restrict__ kh,
                             const _Float16* __restrict__ vt,
                             const int* __restrict__ mask,
                             _Float16* __restrict__ o) {
    __shared__ __align__(128) _Float16 lds_p[4][2 * 16 * 32];

    int wid  = threadIdx.x >> 5;
    int lane = threadIdx.x & 31;
    int gwave = blockIdx.x * (blockDim.x >> 5) + wid;
    const int qstrips = Sc / 32;                 // 64 strips of 32 queries
    int bh = gwave / qstrips;                    // 0..B*H-1
    int qt = gwave - bh * qstrips;
    int b  = bh >> 3;
    int h  = bh & 7;

    const _Float16* qb = qh + (size_t)bh * Sc * Dc;
    const _Float16* kb = kh + (size_t)bh * Sc * Dc;
    const _Float16* vb = vt + (size_t)bh * Dc * Sc;
    const int*      mb = mask + (size_t)b * Sc * Sc;

    int l  = lane & 15;
    int hi = (lane >> 4) & 1;
    int q0 = qt * 32;

    // Q fragments reused across all key iterations (D=64 -> two K-steps)
    v16h qa[2][2];
#pragma unroll
    for (int qi = 0; qi < 2; ++qi) {
        qa[qi][0] = load_frag(qb, Dc, q0 + qi * 16 + l, 0,  lane);
        qa[qi][1] = load_frag(qb, Dc, q0 + qi * 16 + l, 32, lane);
    }

    v8f acc[2][4] = {{{}, {}, {}, {}}, {{}, {}, {}, {}}};
    float mrow[2][8], lrow[2][8];
#pragma unroll
    for (int qi = 0; qi < 2; ++qi)
#pragma unroll
        for (int r = 0; r < 8; ++r) { mrow[qi][r] = -3.0e38f; lrow[qi][r] = 0.0f; }

    _Float16* pbuf = lds_p[wid];

    for (int k0 = 0; k0 < Sc; k0 += 32) {
        // ---- K fragments for this 32-key step: 2 key sub-tiles x 2 d-chunks
        v16h kf[2][2];
#pragma unroll
        for (int t = 0; t < 2; ++t) {
            kf[t][0] = load_frag(kb, Dc, k0 + t * 16 + l, 0,  lane);
            kf[t][1] = load_frag(kb, Dc, k0 + t * 16 + l, 32, lane);
        }

        // ---- scores for both q-tiles (8 WMMAs)
        v8f s[2][2] = {{{}, {}}, {{}, {}}};
#pragma unroll
        for (int qi = 0; qi < 2; ++qi)
#pragma unroll
            for (int t = 0; t < 2; ++t) {
                s[qi][t] = wmma_f16(qa[qi][0], kf[t][0], s[qi][t]);
                s[qi][t] = wmma_f16(qa[qi][1], kf[t][1], s[qi][t]);
            }

        // ---- mask BEFORE scale (reference quirk), online softmax per q-tile
#pragma unroll
        for (int qi = 0; qi < 2; ++qi) {
            const int* mp0 = mb + (size_t)(q0 + qi * 16 + hi * 8) * Sc + k0 + l;
            float p0[8], p1[8];
#pragma unroll
            for (int r = 0; r < 8; ++r) {
                const int* mp = mp0 + (size_t)r * Sc;
                float a = (mp[0]  == 0) ? -1.0e9f : s[qi][0][r];
                float c = (mp[16] == 0) ? -1.0e9f : s[qi][1][r];
                p0[r] = a * 0.125f;
                p1[r] = c * 0.125f;
            }
            float alpha[8];
#pragma unroll
            for (int r = 0; r < 8; ++r) {
                float mx = fmaxf(p0[r], p1[r]);
#pragma unroll
                for (int off = 1; off < 16; off <<= 1)
                    mx = fmaxf(mx, __shfl_xor(mx, off, 32));
                float nm = fmaxf(mrow[qi][r], mx);
                alpha[r] = __expf(mrow[qi][r] - nm);
                mrow[qi][r] = nm;
                float e0 = __expf(p0[r] - nm);
                float e1 = __expf(p1[r] - nm);
                p0[r] = e0; p1[r] = e1;
                float sm = e0 + e1;
#pragma unroll
                for (int off = 1; off < 16; off <<= 1)
                    sm += __shfl_xor(sm, off, 32);
                lrow[qi][r] = lrow[qi][r] * alpha[r] + sm;
            }
#pragma unroll
            for (int dch = 0; dch < 4; ++dch)
#pragma unroll
                for (int r = 0; r < 8; ++r)
                    acc[qi][dch][r] *= alpha[r];

            // C-layout -> A-layout via per-wave LDS tile (lockstep wave)
#pragma unroll
            for (int r = 0; r < 8; ++r) {
                pbuf[qi * 512 + (r + hi * 8) * 32 + l]      = (_Float16)p0[r];
                pbuf[qi * 512 + (r + hi * 8) * 32 + l + 16] = (_Float16)p1[r];
            }
        }

        // ---- load both P fragments and all 4 V fragments up front, then
        //      run the 8-WMMA chain (partial waits, shared V operands)
        v16h pfrag[2];
        pfrag[0] = load_frag(pbuf,       32, l, 0, lane);
        pfrag[1] = load_frag(pbuf + 512, 32, l, 0, lane);
        v16h vfrag[4];
#pragma unroll
        for (int dch = 0; dch < 4; ++dch)
            vfrag[dch] = load_frag(vb, Sc, dch * 16 + l, k0, lane);
#pragma unroll
        for (int dch = 0; dch < 4; ++dch) {
            acc[0][dch] = wmma_f16(pfrag[0], vfrag[dch], acc[0][dch]);
            acc[1][dch] = wmma_f16(pfrag[1], vfrag[dch], acc[1][dch]);
        }
    }

    // ---- normalize, write O as [B,S,H*D] f16 (GEMM-ready, ld=512)
#pragma unroll
    for (int qi = 0; qi < 2; ++qi)
#pragma unroll
        for (int r = 0; r < 8; ++r) {
            float inv = 1.0f / lrow[qi][r];
            int qrow = q0 + qi * 16 + r + hi * 8;
            size_t rowbase = ((size_t)b * Sc + qrow) * (size_t)HDc + h * Dc;
#pragma unroll
            for (int dch = 0; dch < 4; ++dch)
                o[rowbase + dch * 16 + l] = (_Float16)(acc[qi][dch][r] * inv);
        }
}

// ---------------------------------------------------------------------------
extern "C" void kernel_launch(void* const* d_in, const int* in_sizes, int n_in,
                              void* d_out, int out_size, void* d_ws, size_t ws_size,
                              hipStream_t stream) {
    const float* in_q = (const float*)d_in[0];
    const float* in_k = (const float*)d_in[1];
    const float* in_v = (const float*)d_in[2];
    const int*   mask = (const int*)  d_in[3];
    const float* wq   = (const float*)d_in[4];
    const float* bq   = (const float*)d_in[5];
    const float* wk   = (const float*)d_in[6];
    const float* bk   = (const float*)d_in[7];
    const float* wv   = (const float*)d_in[8];
    const float* bv   = (const float*)d_in[9];
    const float* wo   = (const float*)d_in[10];
    const float* bo   = (const float*)d_in[11];
    float* out = (float*)d_out;

    // workspace layout (f16 buffers), all 256B aligned
    const size_t szAct = (size_t)Mc * Ec * sizeof(_Float16);   // 8 MiB each
    const size_t szW   = (size_t)Ec * HDc * sizeof(_Float16);  // 0.5 MiB each
    char* ws = (char*)d_ws;
    _Float16* a16q = (_Float16*)(ws);                    // [M,E]
    _Float16* a16k = (_Float16*)(ws + szAct);
    _Float16* a16v = (_Float16*)(ws + 2 * szAct);
    _Float16* qh   = (_Float16*)(ws + 3 * szAct);        // [B,H,S,D]
    _Float16* kh   = (_Float16*)(ws + 4 * szAct);        // [B,H,S,D]
    _Float16* vt   = (_Float16*)(ws + 5 * szAct);        // [B,H,D,S]
    _Float16* o16  = (_Float16*)(ws + 6 * szAct);        // [M,HD]
    _Float16* wtq  = (_Float16*)(ws + 7 * szAct);        // [HD,E]
    _Float16* wtk  = (_Float16*)(ws + 7 * szAct + szW);
    _Float16* wtv  = (_Float16*)(ws + 7 * szAct + 2 * szW);
    _Float16* wto  = (_Float16*)(ws + 7 * szAct + 3 * szW); // [E,HD]

    // 1) conversions
    {
        int n = Mc * Ec;
        dim3 g((n + 255) / 256), blk(256);
        k_f32_to_f16<<<g, blk, 0, stream>>>(in_q, a16q, n);
        k_f32_to_f16<<<g, blk, 0, stream>>>(in_k, a16k, n);
        k_f32_to_f16<<<g, blk, 0, stream>>>(in_v, a16v, n);
        dim3 gt((HDc + 255) / 256, Ec);
        k_transpose_f16<<<gt, blk, 0, stream>>>(wq, wtq, Ec, HDc);
        k_transpose_f16<<<gt, blk, 0, stream>>>(wk, wtk, Ec, HDc);
        k_transpose_f16<<<gt, blk, 0, stream>>>(wv, wtv, Ec, HDc);
        dim3 gto((Ec + 255) / 256, HDc);
        k_transpose_f16<<<gto, blk, 0, stream>>>(wo, wto, HDc, Ec);
    }

    // 2) projections: (M/16)*(HD/64) = 512*8 = 4096 waves, 8 waves/block
    {
        int waves = (Mc / 16) * (HDc / 64);
        dim3 g(waves / 8), blk(256);
        k_gemm_wmma<<<g, blk, 0, stream>>>(a16q, wtq, bq, qh, nullptr,
                                           Mc, HDc, Ec, 0);
        k_gemm_wmma<<<g, blk, 0, stream>>>(a16k, wtk, bk, kh, nullptr,
                                           Mc, HDc, Ec, 0);
        k_gemm_wmma<<<g, blk, 0, stream>>>(a16v, wtv, bv, vt, nullptr,
                                           Mc, HDc, Ec, 2);
    }

    // 3) flash attention: B*H*(S/32) = 2048 waves, 4 waves/block
    {
        dim3 g((Bc * Hc * (Sc / 32)) / 4), blk(128);
        k_flash_attn<<<g, blk, 0, stream>>>(qh, kh, vt, mask, o16);
    }

    // 4) output projection -> f32
    {
        int waves = (Mc / 16) * (Ec / 64);
        dim3 g(waves / 8), blk(256);
        k_gemm_wmma<<<g, blk, 0, stream>>>(o16, wto, bo, nullptr, out,
                                           Mc, Ec, HDc, 3);
    }
}
